// MultiHeadAttention_66056597013015
// MI455X (gfx1250) — matrix operand order repaired and checked by reading, not compiled, not run on hardware
//
#include <hip/hip_runtime.h>
#include <hip/hip_bf16.h>

#define B_  4
#define T_  2048
#define C_  512
#define H_  8
#define HS_ 64

typedef __bf16 bf16_t;
typedef __attribute__((ext_vector_type(16))) __bf16        v16bf;
typedef __attribute__((ext_vector_type(8)))  float         v8f;
typedef __attribute__((ext_vector_type(4)))  unsigned int  u32x4;

union FragU { v16bf v; u32x4 q[2]; };

__device__ __forceinline__ v8f wmma_bf16(v16bf a, v16bf b, v8f c) {
  // D = A(16x32 bf16) * B(32x16 bf16) + C(16x16 f32)
  return __builtin_amdgcn_wmma_f32_16x16x32_bf16(
      /*neg_a=*/false, a, /*neg_b=*/false, b,
      /*c_mod=*/(short)0, c, /*reuse_a=*/false, /*reuse_b=*/false);
}

// A-fragment: 16x32 bf16 tile from row-major [M][K], stride lda (elements).
// Layout (ISA 7.12.2): lane<16 -> row=lane, elems 0..7 = K0..7, 8..15 = K16..23
//                      lane>=16 -> row=lane-16, elems = K8..15, K24..31
__device__ __forceinline__ v16bf load_a_frag(const bf16_t* p, int lda) {
  int lane = threadIdx.x & 31;
  int m  = lane & 15;
  int k0 = (lane >> 4) << 3;            // 0 or 8
  const bf16_t* s = p + (size_t)m * lda + k0;
  FragU f;
  f.q[0] = *(const u32x4*)(s);          // K k0 .. k0+7
  f.q[1] = *(const u32x4*)(s + 16);     // K k0+16 .. k0+23
  return f.v;
}

// B-fragment: 32x16 bf16 tile, source is "N-major" Bt[n][k] with stride ldb.
// Layout: lane<16 -> col=lane, elems 0..7 = K0..7, 8..15 = K16..23 ; lane>=16 -> col=lane-16, K8..15, K24..31
__device__ __forceinline__ v16bf load_b_frag_t(const bf16_t* p, int ldb) {
  int lane = threadIdx.x & 31;
  int n  = lane & 15;
  int k0 = (lane >> 4) << 3;            // 0 or 8
  const bf16_t* s = p + (size_t)n * ldb + k0;
  FragU f;
  f.q[0] = *(const u32x4*)(s);          // K k0 .. k0+7
  f.q[1] = *(const u32x4*)(s + 16);     // K k0+16 .. k0+23
  return f.v;
}

// Wave-local LDS ordering fence (block barrier would deadlock: waves run
// different key-loop trip counts).
__device__ __forceinline__ void wave_lds_fence() {
  __builtin_amdgcn_wave_barrier();
  asm volatile("s_wait_dscnt 0" ::: "memory");
  __builtin_amdgcn_wave_barrier();
}

// ---------------- conversion / transpose kernels ----------------

__global__ void cvt_f32_bf16(const float* __restrict__ src,
                             bf16_t* __restrict__ dst, int n) {
  int i = blockIdx.x * blockDim.x + threadIdx.x;
  int stride = gridDim.x * blockDim.x;
  for (; i < n; i += stride) dst[i] = (bf16_t)src[i];
}

// W: [batch][rows][cols] f32  ->  WT: [batch][cols][rows] bf16
__global__ void wtrans_f32_bf16(const float* __restrict__ W,
                                bf16_t* __restrict__ WT,
                                int rows, int cols, int n) {
  int i = blockIdx.x * blockDim.x + threadIdx.x;
  int stride = gridDim.x * blockDim.x;
  int per = rows * cols;
  for (; i < n; i += stride) {
    int b = i / per;
    int rem = i - b * per;
    int r = rem / cols;
    int c = rem - r * cols;
    WT[(size_t)b * per + (size_t)c * rows + r] = (bf16_t)W[i];
  }
}

// ---------------- QKV projection GEMM ----------------
// x[b] (T x C) @ W[h] (C x HS) for q/k/v.  Wave computes 32x64 output tile.
// grid = (T/128, B*H, 3), block = 128 (4 waves).
__global__ __launch_bounds__(128)
void qkv_gemm(const bf16_t* __restrict__ xb,
              const bf16_t* __restrict__ wqT, const bf16_t* __restrict__ wkT,
              const bf16_t* __restrict__ wvT,
              bf16_t* __restrict__ q, bf16_t* __restrict__ k,
              bf16_t* __restrict__ vT) {
  int w    = threadIdx.x >> 5;
  int lane = threadIdx.x & 31;
  int col16 = lane & 15;
  int hi8   = (lane >> 4) << 3;
  int bh = blockIdx.y;
  int h  = bh % H_;
  int sel = blockIdx.z;
  int b  = bh / H_;
  int mbase = blockIdx.x * 128 + w * 32;

  const bf16_t* A  = xb + (size_t)b * T_ * C_;
  const bf16_t* WT = (sel == 0 ? wqT : (sel == 1 ? wkT : wvT)) + (size_t)h * HS_ * C_;

  v8f zero = {};
  v8f acc[2][4];
#pragma unroll
  for (int mt = 0; mt < 2; ++mt)
#pragma unroll
    for (int nt = 0; nt < 4; ++nt) acc[mt][nt] = zero;

  for (int kk = 0; kk < C_ / 32; ++kk) {
    v16bf a0 = load_a_frag(A + (size_t)mbase * C_ + kk * 32, C_);
    v16bf a1 = load_a_frag(A + (size_t)(mbase + 16) * C_ + kk * 32, C_);
#pragma unroll
    for (int nt = 0; nt < 4; ++nt) {
      v16bf bf = load_b_frag_t(WT + (size_t)(nt * 16) * C_ + kk * 32, C_);
      acc[0][nt] = wmma_bf16(a0, bf, acc[0][nt]);
      acc[1][nt] = wmma_bf16(a1, bf, acc[1][nt]);
    }
  }

  if (sel < 2) {
    bf16_t* out = (sel == 0 ? q : k) + (size_t)bh * T_ * HS_;
#pragma unroll
    for (int mt = 0; mt < 2; ++mt)
#pragma unroll
      for (int nt = 0; nt < 4; ++nt)
#pragma unroll
        for (int r = 0; r < 8; ++r) {
          int row = mbase + mt * 16 + r + hi8;
          out[(size_t)row * HS_ + nt * 16 + col16] = (bf16_t)acc[mt][nt][r];
        }
  } else {
    bf16_t* out = vT + (size_t)bh * HS_ * T_;   // [HS][T]
#pragma unroll
    for (int mt = 0; mt < 2; ++mt)
#pragma unroll
      for (int nt = 0; nt < 4; ++nt)
#pragma unroll
        for (int r = 0; r < 8; ++r) {
          int row = mbase + mt * 16 + r + hi8;
          out[(size_t)(nt * 16 + col16) * T_ + row] = (bf16_t)acc[mt][nt][r];
        }
  }
}

// ---------------- flash attention (causal, online softmax) ----------------
// grid = (T/128, H, B), block = 128 (4 waves); wave owns 32 query rows.
__global__ __launch_bounds__(128)
void attn_flash(const bf16_t* __restrict__ q, const bf16_t* __restrict__ k,
                const bf16_t* __restrict__ vT, bf16_t* __restrict__ heads) {
  __shared__ bf16_t pbuf[4][32][32];   // per-wave P tile (bf16)

  int w    = threadIdx.x >> 5;
  int lane = threadIdx.x & 31;
  int col16 = lane & 15;
  int hi8   = (lane >> 4) << 3;
  int h = blockIdx.y, b = blockIdx.z;
  int bh = b * H_ + h;
  int mbase = blockIdx.x * 128 + w * 32;

  const bf16_t* qb = q  + (size_t)bh * T_ * HS_;
  const bf16_t* kb = k  + (size_t)bh * T_ * HS_;
  const bf16_t* vb = vT + (size_t)bh * HS_ * T_;

  v16bf qf[2][2];
#pragma unroll
  for (int mt = 0; mt < 2; ++mt)
#pragma unroll
    for (int kk = 0; kk < 2; ++kk)
      qf[mt][kk] = load_a_frag(qb + (size_t)(mbase + mt * 16) * HS_ + kk * 32, HS_);

  v8f zero = {};
  v8f acc[2][4];
  float mst[2][8], lst[2][8];
#pragma unroll
  for (int mt = 0; mt < 2; ++mt) {
#pragma unroll
    for (int nt = 0; nt < 4; ++nt) acc[mt][nt] = zero;
#pragma unroll
    for (int r = 0; r < 8; ++r) { mst[mt][r] = -__builtin_inff(); lst[mt][r] = 0.0f; }
  }

  const float scale = 0.125f;  // HS^-0.5

  for (int sb = 0; sb <= mbase; sb += 32) {
    // ---- scores: S = (Q Kt) * scale, causal mask ----
    v8f sc[2][2];
#pragma unroll
    for (int mt = 0; mt < 2; ++mt)
#pragma unroll
      for (int st = 0; st < 2; ++st) {
        v8f s = zero;
#pragma unroll
        for (int kk = 0; kk < 2; ++kk) {
          v16bf bf = load_b_frag_t(kb + (size_t)(sb + st * 16) * HS_ + kk * 32, HS_);
          s = wmma_bf16(qf[mt][kk], bf, s);
        }
        int col = sb + st * 16 + col16;
#pragma unroll
        for (int r = 0; r < 8; ++r) {
          int row = mbase + mt * 16 + r + hi8;
          s[r] = (col <= row) ? s[r] * scale : -__builtin_inff();
        }
        sc[mt][st] = s;
      }

    // ---- online softmax update (row-wise over half-wave) ----
#pragma unroll
    for (int mt = 0; mt < 2; ++mt) {
#pragma unroll
      for (int r = 0; r < 8; ++r) {
        float tm = fmaxf(sc[mt][0][r], sc[mt][1][r]);
        tm = fmaxf(tm, __shfl_xor(tm, 1));
        tm = fmaxf(tm, __shfl_xor(tm, 2));
        tm = fmaxf(tm, __shfl_xor(tm, 4));
        tm = fmaxf(tm, __shfl_xor(tm, 8));
        float mold = mst[mt][r];
        float mnew = fmaxf(mold, tm);
        float alpha = __expf(mold - mnew);
        float p0 = __expf(sc[mt][0][r] - mnew);
        float p1 = __expf(sc[mt][1][r] - mnew);
        pbuf[w][mt * 16 + r + hi8][col16]      = (bf16_t)p0;
        pbuf[w][mt * 16 + r + hi8][16 + col16] = (bf16_t)p1;
        float ps = p0 + p1;
        ps += __shfl_xor(ps, 1);
        ps += __shfl_xor(ps, 2);
        ps += __shfl_xor(ps, 4);
        ps += __shfl_xor(ps, 8);
        lst[mt][r] = lst[mt][r] * alpha + ps;
        mst[mt][r] = mnew;
#pragma unroll
        for (int nt = 0; nt < 4; ++nt) acc[mt][nt][r] *= alpha;
      }
    }

    wave_lds_fence();

    // ---- O += P @ V  (B-fragments from transposed V: contiguous per lane) ----
    v16bf pf0 = load_a_frag(&pbuf[w][0][0], 32);
    v16bf pf1 = load_a_frag(&pbuf[w][16][0], 32);
#pragma unroll
    for (int nt = 0; nt < 4; ++nt) {
      v16bf bf = load_b_frag_t(vb + (size_t)(nt * 16) * T_ + sb, T_);
      acc[0][nt] = wmma_bf16(pf0, bf, acc[0][nt]);
      acc[1][nt] = wmma_bf16(pf1, bf, acc[1][nt]);
    }

    wave_lds_fence();
  }

  // ---- normalize and store head outputs (bf16, [B,T,H*HS]) ----
#pragma unroll
  for (int mt = 0; mt < 2; ++mt)
#pragma unroll
    for (int r = 0; r < 8; ++r) {
      float inv = 1.0f / lst[mt][r];
      int t = mbase + mt * 16 + r + hi8;
#pragma unroll
      for (int nt = 0; nt < 4; ++nt)
        heads[((size_t)b * T_ + t) * C_ + h * HS_ + nt * 16 + col16] =
            (bf16_t)(acc[mt][nt][r] * inv);
    }
}

// ---------------- output projection GEMM ----------------
// heads (B*T x C) @ Wp (C x C) -> f32 out. grid = (B*T/128, C/64), block 128.
__global__ __launch_bounds__(128)
void proj_gemm(const bf16_t* __restrict__ heads,
               const bf16_t* __restrict__ wpT, float* __restrict__ out) {
  int w    = threadIdx.x >> 5;
  int lane = threadIdx.x & 31;
  int col16 = lane & 15;
  int hi8   = (lane >> 4) << 3;
  int mbase = blockIdx.x * 128 + w * 32;
  int nbase = blockIdx.y * 64;

  v8f zero = {};
  v8f acc[2][4];
#pragma unroll
  for (int mt = 0; mt < 2; ++mt)
#pragma unroll
    for (int nt = 0; nt < 4; ++nt) acc[mt][nt] = zero;

  for (int kk = 0; kk < C_ / 32; ++kk) {
    v16bf a0 = load_a_frag(heads + (size_t)mbase * C_ + kk * 32, C_);
    v16bf a1 = load_a_frag(heads + (size_t)(mbase + 16) * C_ + kk * 32, C_);
#pragma unroll
    for (int nt = 0; nt < 4; ++nt) {
      v16bf bf = load_b_frag_t(wpT + (size_t)(nbase + nt * 16) * C_ + kk * 32, C_);
      acc[0][nt] = wmma_bf16(a0, bf, acc[0][nt]);
      acc[1][nt] = wmma_bf16(a1, bf, acc[1][nt]);
    }
  }

#pragma unroll
  for (int mt = 0; mt < 2; ++mt)
#pragma unroll
    for (int nt = 0; nt < 4; ++nt)
#pragma unroll
      for (int r = 0; r < 8; ++r) {
        int row = mbase + mt * 16 + r + hi8;
        out[(size_t)row * C_ + nbase + nt * 16 + col16] = acc[mt][nt][r];
      }
}

// ---------------- host launch ----------------

extern "C" void kernel_launch(void* const* d_in, const int* in_sizes, int n_in,
                              void* d_out, int out_size, void* d_ws, size_t ws_size,
                              hipStream_t stream) {
  (void)in_sizes; (void)n_in; (void)out_size; (void)ws_size;
  const float* x  = (const float*)d_in[0];
  const float* Wq = (const float*)d_in[1];
  const float* Wk = (const float*)d_in[2];
  const float* Wv = (const float*)d_in[3];
  const float* Wp = (const float*)d_in[4];
  float* out = (float*)d_out;

  char* ws = (char*)d_ws;
  auto alloc = [&](size_t bytes) {
    void* p = (void*)ws;
    ws += (bytes + 255) & ~(size_t)255;
    return p;
  };
  bf16_t* xb    = (bf16_t*)alloc((size_t)B_ * T_ * C_ * 2);
  bf16_t* wqT   = (bf16_t*)alloc((size_t)H_ * HS_ * C_ * 2);
  bf16_t* wkT   = (bf16_t*)alloc((size_t)H_ * HS_ * C_ * 2);
  bf16_t* wvT   = (bf16_t*)alloc((size_t)H_ * HS_ * C_ * 2);
  bf16_t* wpT   = (bf16_t*)alloc((size_t)C_ * C_ * 2);
  bf16_t* qbuf  = (bf16_t*)alloc((size_t)B_ * H_ * T_ * HS_ * 2);
  bf16_t* kbuf  = (bf16_t*)alloc((size_t)B_ * H_ * T_ * HS_ * 2);
  bf16_t* vTbuf = (bf16_t*)alloc((size_t)B_ * H_ * T_ * HS_ * 2);
  bf16_t* heads = (bf16_t*)alloc((size_t)B_ * T_ * C_ * 2);

  int nx = B_ * T_ * C_;
  cvt_f32_bf16<<<(nx + 255) / 256, 256, 0, stream>>>(x, xb, nx);

  int nw = H_ * C_ * HS_;
  wtrans_f32_bf16<<<(nw + 255) / 256, 256, 0, stream>>>(Wq, wqT, C_, HS_, nw);
  wtrans_f32_bf16<<<(nw + 255) / 256, 256, 0, stream>>>(Wk, wkT, C_, HS_, nw);
  wtrans_f32_bf16<<<(nw + 255) / 256, 256, 0, stream>>>(Wv, wvT, C_, HS_, nw);
  int np = C_ * C_;
  wtrans_f32_bf16<<<(np + 255) / 256, 256, 0, stream>>>(Wp, wpT, C_, C_, np);

  qkv_gemm<<<dim3(T_ / 128, B_ * H_, 3), 128, 0, stream>>>(
      xb, wqT, wkT, wvT, qbuf, kbuf, vTbuf);

  attn_flash<<<dim3(T_ / 128, H_, B_), 128, 0, stream>>>(
      qbuf, kbuf, vTbuf, heads);

  proj_gemm<<<dim3(B_ * T_ / 128, C_ / 64), 128, 0, stream>>>(
      heads, wpT, out);
}
